// AttentionHead_11390253269612
// MI455X (gfx1250) — matrix operand order repaired and checked by reading, not compiled, not run on hardware
//
#include <hip/hip_runtime.h>
#include <hip/hip_bf16.h>

typedef __attribute__((ext_vector_type(16))) _Float16 v16h;
typedef __attribute__((ext_vector_type(8)))  _Float16 v8h;
typedef __attribute__((ext_vector_type(8)))  float    v8f;
typedef __attribute__((ext_vector_type(4)))  float    v4f;
typedef __attribute__((ext_vector_type(4)))  int      v4i;

#define B_ 4
#define S_ 4096
#define D_ 1024
#define H_ 64

#define AS1 __attribute__((address_space(1)))
#define AS3 __attribute__((address_space(3)))

#if defined(__AMDGCN__) && __has_builtin(__builtin_amdgcn_global_load_async_to_lds_b128)
#define ASYNC_LDS 1
#else
#define ASYNC_LDS 0
#endif

#if defined(__AMDGCN__) && __has_builtin(__builtin_amdgcn_s_wait_asynccnt)
#define WAIT_ASYNC(n) __builtin_amdgcn_s_wait_asynccnt(n)
#elif ASYNC_LDS
#define WAIT_ASYNC(n) asm volatile("s_wait_asynccnt %0" ::"n"(n) : "memory")
#else
#define WAIT_ASYNC(n) ((void)0)
#endif

union V16U { v16h v; v8h h[2]; };

// ---------------------------------------------------------------------------
// Kernel 0: transpose + convert the three weight matrices once.
//   W[D,H] f32  ->  wt[3][H][D] f16   (384 KB, L2-resident)
// ---------------------------------------------------------------------------
__global__ __launch_bounds__(256) void prep_w_kernel(
    const float* __restrict__ Wq, const float* __restrict__ Wk,
    const float* __restrict__ Wv, _Float16* __restrict__ wt)
{
    const int idx = blockIdx.x * 256 + threadIdx.x;   // 3*64*128 threads
    const int w   = idx / (H_ * (D_ / 8));
    const int r   = idx % (H_ * (D_ / 8));
    const int n   = r / (D_ / 8);
    const int d8  = (r % (D_ / 8)) * 8;
    const float* W = (w == 0) ? Wq : (w == 1) ? Wk : Wv;
    v8h o;
    #pragma unroll
    for (int j = 0; j < 8; ++j) o[j] = (_Float16)W[(long)(d8 + j) * H_ + n];
    *(v8h*)(wt + ((long)w * H_ + n) * D_ + d8) = o;
}

// ---------------------------------------------------------------------------
// Kernel 1: fused QKV projection.  One wave: 32-row x 64-col tile for all
// three outputs.  q,k row-major f16; v stored transposed vT[B][H][S] f16.
// Per k-step: 8 v4f (A) + 24 b128 (B) loads -> 24 v_wmma_f32_16x16x32_f16.
// ---------------------------------------------------------------------------
__global__ __launch_bounds__(256) void qkv_proj_kernel(
    const float* __restrict__ x, const _Float16* __restrict__ wt,
    _Float16* __restrict__ q, _Float16* __restrict__ k, _Float16* __restrict__ vT)
{
    const int lane = threadIdx.x & 31;
    const int wave = threadIdx.x >> 5;
    const int tile = blockIdx.x * 8 + wave;          // 32-row tile of M = B*S

    const int half_sel = lane >> 4;
    const int l16      = lane & 15;
    const int aoff     = half_sel * 8;

    v8f acc[3][2][4];
    #pragma unroll
    for (int w = 0; w < 3; ++w)
        #pragma unroll
        for (int m = 0; m < 2; ++m)
            #pragma unroll
            for (int t = 0; t < 4; ++t) acc[w][m][t] = v8f{};

    for (int kk = 0; kk < D_; kk += 32) {
        v16h a[2];
        #pragma unroll
        for (int m = 0; m < 2; ++m) {
            const long arow = (long)tile * 32 + m * 16 + l16;
            const float* xa = x + arow * D_ + kk + aoff;
            v4f f0 = *(const v4f*)(xa + 0);
            v4f f1 = *(const v4f*)(xa + 4);
            v4f f2 = *(const v4f*)(xa + 16);
            v4f f3 = *(const v4f*)(xa + 20);
            #pragma unroll
            for (int j = 0; j < 4; ++j) {
                a[m][j]      = (_Float16)f0[j];
                a[m][4 + j]  = (_Float16)f1[j];
                a[m][8 + j]  = (_Float16)f2[j];
                a[m][12 + j] = (_Float16)f3[j];
            }
        }
        #pragma unroll
        for (int w = 0; w < 3; ++w) {
            #pragma unroll
            for (int t = 0; t < 4; ++t) {
                const _Float16* p = wt + ((long)w * H_ + t * 16 + l16) * D_
                                       + kk + half_sel * 8;
                V16U u;
                u.h[0] = *(const v8h*)(p + 0);
                u.h[1] = *(const v8h*)(p + 16);
                #pragma unroll
                for (int m = 0; m < 2; ++m)
                    acc[w][m][t] = __builtin_amdgcn_wmma_f32_16x16x32_f16(
                        false, a[m], false, u.v, (short)0, acc[w][m][t], false, false);
            }
        }
    }

    #pragma unroll
    for (int m = 0; m < 2; ++m) {
        #pragma unroll
        for (int t = 0; t < 4; ++t) {
            #pragma unroll
            for (int j = 0; j < 8; ++j) {
                const long row = (long)tile * 32 + m * 16 + j + half_sel * 8;
                const int  col = t * 16 + l16;
                q[row * H_ + col] = (_Float16)acc[0][m][t][j];
                k[row * H_ + col] = (_Float16)acc[1][m][t][j];
                const long b = row >> 12;
                const long s = row & (S_ - 1);
                vT[(b * H_ + col) * S_ + s] = (_Float16)acc[2][m][t][j];
            }
        }
    }
}

// ---------------------------------------------------------------------------
// Kernel 2: flash attention, 64 keys/iteration.
//  - K tiles (8 KB, contiguous) streamed into per-wave LDS via
//    GLOBAL_LOAD_ASYNC_TO_LDS_B128, double-buffered, drained with
//    s_wait_asynccnt <= 16 so next tile overlaps current compute.
//  - V fragments: contiguous b128 from transposed vT.
//  - f32 online softmax (half-wave shfl_xor reductions match C-layout rows),
//    P (16x64) round-trips per-wave LDS into two A-fragments.
//  16 v_wmma per iteration.  2 waves/block, 36 KB LDS/block.
// ---------------------------------------------------------------------------
__global__ __launch_bounds__(64) void flash_attn_kernel(
    const _Float16* __restrict__ q,
    const _Float16* __restrict__ k,
    const _Float16* __restrict__ vT,
    float* __restrict__ out)
{
    __shared__ _Float16 kst[2][2][64 * 64];          // [wave][dbuf] 8 KB tiles
    __shared__ _Float16 plds[2][16 * 64];            // P staging, 2 KB/wave

    const int lane = threadIdx.x & 31;
    const int wave = threadIdx.x >> 5;               // 0..1
    const int tile = blockIdx.x * 2 + wave;          // 0 .. B*S/16-1
    const int b    = tile >> 8;                      // 256 q-tiles per batch
    const int qt   = tile & 255;

    const int half_sel = lane >> 4;
    const int l16      = lane & 15;

    const _Float16* qb  = q  + (long)b * S_ * H_;
    const _Float16* kb  = k  + (long)b * S_ * H_;
    const _Float16* vtb = vT + (long)b * H_ * S_;
    _Float16* lds = plds[wave];

    // ---- async-stage one 64-key K tile (8 KB contiguous) into LDS ----
    auto issueK = [&](int kt, int buf) {
        const _Float16* gsrc = kb + (long)kt * 64 * H_;
        _Float16*       ldst = &kst[wave][buf][0];
        #pragma unroll
        for (int i = 0; i < 16; ++i) {
            const int off = (i * 32 + lane) * 8;     // 16-byte chunks
#if ASYNC_LDS
            __builtin_amdgcn_global_load_async_to_lds_b128(
                (AS1 v4i*)(gsrc + off), (AS3 v4i*)(ldst + off), 0, 0);
#else
            *(v8h*)(ldst + off) = *(const v8h*)(gsrc + off);
#endif
        }
    };

    // ---- Q A-fragments (16x64 -> two 16x32 fragments), loaded once ----
    v16h aq[2];
    {
        const long arow = (long)qt * 16 + l16;
        #pragma unroll
        for (int c = 0; c < 2; ++c) {
            const _Float16* p = qb + arow * H_ + c * 32 + half_sel * 8;
            V16U u;
            u.h[0] = *(const v8h*)(p + 0);
            u.h[1] = *(const v8h*)(p + 16);
            aq[c] = u.v;
        }
    }

    const float scale = 0.125f;                      // H^-0.5

    float m_r[8], l_r[8];
    #pragma unroll
    for (int j = 0; j < 8; ++j) { m_r[j] = -1e30f; l_r[j] = 0.0f; }
    v8f oacc[4] = {v8f{}, v8f{}, v8f{}, v8f{}};

    const int ntiles = S_ / 64;                      // 64 iterations of 64 keys
    issueK(0, 0);

    for (int kt = 0; kt < ntiles; ++kt) {
        const int keybase = kt * 64;
        const int cb = kt & 1;

        if (kt + 1 < ntiles) {
            issueK(kt + 1, cb ^ 1);                  // overlap next K tile
            WAIT_ASYNC(16);                          // current tile complete
            __builtin_prefetch(vtb + (long)lane * S_ + keybase + 64, 0, 3);
            __builtin_prefetch(vtb + (long)(32 + lane) * S_ + keybase + 64, 0, 3);
        } else {
            WAIT_ASYNC(0);
        }
        asm volatile("s_wait_dscnt 0x0" ::: "memory");  // fallback-path ordering

        // ---- S = Q . K^T : four 16x16 accumulators from LDS K tile ----
        v8f sacc[4];
        #pragma unroll
        for (int s = 0; s < 4; ++s) {
            sacc[s] = v8f{};
            const _Float16* krow = &kst[wave][cb][(s * 16 + l16) * H_];
            #pragma unroll
            for (int c = 0; c < 2; ++c) {
                const _Float16* p = krow + c * 32 + half_sel * 8;
                V16U u;
                u.h[0] = *(const v8h*)(p + 0);
                u.h[1] = *(const v8h*)(p + 16);
                sacc[s] = __builtin_amdgcn_wmma_f32_16x16x32_f16(
                    false, aq[c], false, u.v, (short)0, sacc[s], false, false);
            }
        }

        // ---- online softmax per row; write P (f16) to LDS row-major ----
        #pragma unroll
        for (int j = 0; j < 8; ++j) {
            float s0 = sacc[0][j] * scale;
            float s1 = sacc[1][j] * scale;
            float s2 = sacc[2][j] * scale;
            float s3 = sacc[3][j] * scale;
            float rmax = fmaxf(fmaxf(s0, s1), fmaxf(s2, s3));
            #pragma unroll
            for (int w = 1; w < 16; w <<= 1)
                rmax = fmaxf(rmax, __shfl_xor(rmax, w, 32));
            float mnew  = fmaxf(m_r[j], rmax);
            float alpha = __expf(m_r[j] - mnew);
            float p0 = __expf(s0 - mnew);
            float p1 = __expf(s1 - mnew);
            float p2 = __expf(s2 - mnew);
            float p3 = __expf(s3 - mnew);
            float rsum = (p0 + p1) + (p2 + p3);
            #pragma unroll
            for (int w = 1; w < 16; w <<= 1)
                rsum += __shfl_xor(rsum, w, 32);
            l_r[j] = l_r[j] * alpha + rsum;
            m_r[j] = mnew;
            #pragma unroll
            for (int t = 0; t < 4; ++t) oacc[t][j] *= alpha;

            const int row = j + half_sel * 8;
            lds[row * 64 +      l16] = (_Float16)p0;
            lds[row * 64 + 16 + l16] = (_Float16)p1;
            lds[row * 64 + 32 + l16] = (_Float16)p2;
            lds[row * 64 + 48 + l16] = (_Float16)p3;
        }

        asm volatile("s_wait_dscnt 0x0" ::: "memory");  // cross-lane P handoff

        // ---- P A-fragments (16x64 -> two 16x32) from LDS ----
        V16U up[2];
        #pragma unroll
        for (int c2 = 0; c2 < 2; ++c2) {
            const _Float16* pr = lds + l16 * 64 + c2 * 32 + half_sel * 8;
            up[c2].h[0] = *(const v8h*)(pr + 0);
            up[c2].h[1] = *(const v8h*)(pr + 16);
        }

        // ---- O += P . V : 4 N-tiles x 2 k-chunks, contiguous b128 from vT ----
        #pragma unroll
        for (int t = 0; t < 4; ++t) {
            const _Float16* vrow = vtb + (long)(t * 16 + l16) * S_ + keybase;
            #pragma unroll
            for (int c2 = 0; c2 < 2; ++c2) {
                const _Float16* p = vrow + c2 * 32 + half_sel * 8;
                V16U u;
                u.h[0] = *(const v8h*)(p + 0);
                u.h[1] = *(const v8h*)(p + 16);
                oacc[t] = __builtin_amdgcn_wmma_f32_16x16x32_f16(
                    false, up[c2].v, false, u.v, (short)0, oacc[t], false, false);
            }
        }
    }

    // ---- epilogue: O / l, write f32 output [B,S,H] ----
    #pragma unroll
    for (int t = 0; t < 4; ++t) {
        #pragma unroll
        for (int j = 0; j < 8; ++j) {
            const long row = (long)b * S_ + (long)qt * 16 + j + half_sel * 8;
            out[row * H_ + t * 16 + l16] = oacc[t][j] / l_r[j];
        }
    }
}

// ---------------------------------------------------------------------------
extern "C" void kernel_launch(void* const* d_in, const int* in_sizes, int n_in,
                              void* d_out, int out_size, void* d_ws, size_t ws_size,
                              hipStream_t stream) {
    (void)in_sizes; (void)n_in; (void)out_size; (void)ws_size;
    const float* x  = (const float*)d_in[0];
    const float* Wq = (const float*)d_in[1];
    const float* Wk = (const float*)d_in[2];
    const float* Wv = (const float*)d_in[3];
    float* out = (float*)d_out;

    const size_t nW = (size_t)3 * H_ * D_;           // 196608 halves
    const size_t nT = (size_t)B_ * S_ * H_;          // 1M halves per tensor
    _Float16* wt = (_Float16*)d_ws;
    _Float16* qw = wt + nW;
    _Float16* kw = qw + nT;
    _Float16* vt = kw + nT;

    prep_w_kernel<<<dim3((3 * H_ * (D_ / 8)) / 256), 256, 0, stream>>>(Wq, Wk, Wv, wt);
    qkv_proj_kernel<<<dim3(64), 256, 0, stream>>>(x, wt, qw, kw, vt);
    flash_attn_kernel<<<dim3(512), 64, 0, stream>>>(qw, kw, vt, out);
}
